// DotAttention_26860725469361
// MI455X (gfx1250) — matrix operand order repaired and checked by reading, not compiled, not run on hardware
//
#include <hip/hip_runtime.h>

// DotAttention for MI455X (gfx1250).
// Pass 0: pre-convert Q,K -> bf16 (pair-packed), V -> bf16 transposed tiles (ws).
// Pass 1: exp(Q K^T) via v_wmma_f32_16x16x32_bf16, async global->LDS tile copies,
//         deterministic partial row sums.
// Pass 2: row-sum reduce -> reciprocal.
// Pass 3: normalize weights in place + x = P*V (bf16 WMMA).
// d_out = [x (16384x256 f32) | atten_weight (16384x16384 f32)].

#define NQ  16384
#define NK  16384
#define DD  256
#define DVV 256

typedef __attribute__((ext_vector_type(16))) __bf16       v16bf;
typedef __attribute__((ext_vector_type(8)))  float        v8f;
typedef __attribute__((ext_vector_type(8)))  unsigned int v8u;
typedef unsigned int u32;
typedef __attribute__((address_space(3))) u32 lds_u32_t;

__device__ __forceinline__ u32 pack_bf16(float a, float b) {
  // round-to-nearest-even f32 -> bf16, packed pair (a = low half)
  u32 ua = __builtin_bit_cast(u32, a);
  u32 ub = __builtin_bit_cast(u32, b);
  ua += 0x7FFFu + ((ua >> 16) & 1u);
  ub += 0x7FFFu + ((ub >> 16) & 1u);
  return (ua >> 16) | (ub & 0xFFFF0000u);
}

// async 16B global -> LDS copy (per lane), tracked with ASYNCcnt
__device__ __forceinline__ void async_b128(u32* lds_ptr, const u32* gptr) {
  u32 off = (u32)(unsigned long long)(lds_u32_t*)lds_ptr;   // LDS byte offset
  asm volatile("global_load_async_to_lds_b128 %0, %1, off"
               :: "v"(off), "v"(gptr) : "memory");
}
__device__ __forceinline__ void async_wait0() {
  asm volatile("s_wait_asynccnt 0x0" ::: "memory");
}

// ---------------------------------------------------------------------------
// Pass 0a: pack pairs of f32 -> one dword of 2 x bf16 (row-major, used for Q,K)
// ---------------------------------------------------------------------------
__global__ __launch_bounds__(256) void pack_rows_kernel(
    const float* __restrict__ src, u32* __restrict__ dst)
{
  const int idx = blockIdx.x * 256 + threadIdx.x;        // one dword per thread
  const float2 f = ((const float2*)src)[idx];
  dst[idx] = pack_bf16(f.x, f.y);
}

// ---------------------------------------------------------------------------
// Pass 0b: V (16384x256 f32) -> VT tiles: block b holds k-rows 32b..32b+31,
// VT[((b*256)+col)*16 + p] packs V[32b+2p][col], V[32b+2p+1][col].
// This is exactly the bf16 WMMA B-fragment pairing -> pure copy at use site.
// ---------------------------------------------------------------------------
__global__ __launch_bounds__(256) void pack_vt_kernel(
    const float* __restrict__ V, u32* __restrict__ VT)
{
  const int b = blockIdx.x;          // 0..511 (32-row blocks)
  const int c = threadIdx.x;         // column 0..255
#pragma unroll
  for (int p = 0; p < 16; ++p) {
    const float lo = V[(size_t)(32 * b + 2 * p)     * DVV + c];
    const float hi = V[(size_t)(32 * b + 2 * p + 1) * DVV + c];
    VT[((size_t)b * 256 + c) * 16 + p] = pack_bf16(lo, hi);
  }
}

// ---------------------------------------------------------------------------
// Pass 1: 128x128 tile of exp(Q K^T); partial row sums per k-tile (no atomics)
// grid (NK/128, NQ/128), block 256 (8 waves).  QB/KB: bf16 pair-packed rows.
// ---------------------------------------------------------------------------
__global__ __launch_bounds__(256) void scores_kernel(
    const u32* __restrict__ QB, const u32* __restrict__ KB,
    float* __restrict__ attn, float* __restrict__ partial)
{
  __shared__ u32 qs[128 * 20];   // 128 rows x 16 dwords, stride 20 (16B aligned)
  __shared__ u32 ks[128 * 20];

  const int kblock = blockIdx.x * 128;
  const int qblock = blockIdx.y * 128;
  const int tid  = threadIdx.x;
  const int wv   = tid >> 5;          // wave 0..7: rows 16*wv..+16
  const int lane = tid & 31;
  const int l = lane & 15, h = lane >> 4;

  const int lrow  = tid >> 1;         // loader: row 0..127
  const int lhalf = tid & 1;          // loader: 8-dword half

  const v8f vzero = {0.f,0.f,0.f,0.f,0.f,0.f,0.f,0.f};
  v8f acc[8];
#pragma unroll
  for (int t = 0; t < 8; ++t) acc[t] = vzero;

  for (int kd = 0; kd < DD; kd += 32) {
    {
      const u32* gq = QB + (size_t)(qblock + lrow) * (DD / 2) + (kd >> 1) + lhalf * 8;
      const u32* gk = KB + (size_t)(kblock + lrow) * (DD / 2) + (kd >> 1) + lhalf * 8;
      u32* dq = qs + lrow * 20 + lhalf * 8;
      u32* dk = ks + lrow * 20 + lhalf * 8;
      async_b128(dq,     gq);
      async_b128(dq + 4, gq + 4);
      async_b128(dk,     gk);
      async_b128(dk + 4, gk + 4);
    }
    async_wait0();
    __syncthreads();

    // A fragment: 16x32 bf16, row M=l, K pairs per dword; lanes 16-31 hold K=16..31
    v8u au;
    const u32* ap = qs + (16 * wv + l) * 20 + h * 8;
    au[0] = ap[0]; au[1] = ap[1]; au[2] = ap[2]; au[3] = ap[3];
    au[4] = ap[4]; au[5] = ap[5]; au[6] = ap[6]; au[7] = ap[7];
    v16bf a = __builtin_bit_cast(v16bf, au);

#pragma unroll
    for (int t = 0; t < 8; ++t) {
      // B fragment: B[k][n] = K[n][k]; lane = n, lanes 16-31 hold K=16..31
      const u32* bp = ks + (16 * t + l) * 20 + h * 8;
      v8u bu;
      bu[0] = bp[0]; bu[1] = bp[1]; bu[2] = bp[2]; bu[3] = bp[3];
      bu[4] = bp[4]; bu[5] = bp[5]; bu[6] = bp[6]; bu[7] = bp[7];
      v16bf b = __builtin_bit_cast(v16bf, bu);
      acc[t] = __builtin_amdgcn_wmma_f32_16x16x32_bf16(
          false, a, false, b, (short)0, acc[t], false, false);
    }
    __syncthreads();
  }

  // epilogue: exp(S) (range-safe: max score ~70 < 88), store raw, partial sums
  float psum[8] = {0.f,0.f,0.f,0.f,0.f,0.f,0.f,0.f};
#pragma unroll
  for (int t = 0; t < 8; ++t) {
#pragma unroll
    for (int r = 0; r < 8; ++r) {
      float e = __expf(acc[t][r]);
      const int qrow = qblock + 16 * wv + 8 * h + r;   // C/D layout: M = 8h+r
      const int col  = kblock + 16 * t + l;            // N = l
      attn[(size_t)qrow * NK + col] = e;
      psum[r] += e;
    }
  }
#pragma unroll
  for (int r = 0; r < 8; ++r) {
    float s = psum[r];
    s += __shfl_xor(s, 1);   // xor masks < 16 stay within each 16-lane half
    s += __shfl_xor(s, 2);
    s += __shfl_xor(s, 4);
    s += __shfl_xor(s, 8);
    if (l == r)
      partial[(size_t)blockIdx.x * NQ + qblock + 16 * wv + 8 * h + r] = s;
  }
}

// ---------------------------------------------------------------------------
// Pass 2: reduce 128 partials per row -> reciprocal row sum
// ---------------------------------------------------------------------------
__global__ __launch_bounds__(256) void rowsum_kernel(
    const float* __restrict__ partial, float* __restrict__ rinv)
{
  const int q = blockIdx.x * 256 + threadIdx.x;
  float s = 0.f;
  for (int kb = 0; kb < NK / 128; ++kb)
    s += partial[(size_t)kb * NQ + q];
  rinv[q] = 1.0f / s;
}

// ---------------------------------------------------------------------------
// Pass 3: normalize P in place + x = P*V.  grid NQ/128, block 512 (16 waves)
// wave wv: rows 16*(wv&7), column half 128*(wv>>3)
// ---------------------------------------------------------------------------
__global__ __launch_bounds__(512) void pv_kernel(
    const u32* __restrict__ VT, float* __restrict__ attn,
    const float* __restrict__ rinv, float* __restrict__ X)
{
  __shared__ u32 ps[128 * 17];    // P tile 128 x 32 bf16 (row-major pairs)
  __shared__ u32 vsT[256 * 20];   // V tile: [col][k-pair], stride 20 (16B align)

  const int qblock = blockIdx.x * 128;
  const int tid  = threadIdx.x;
  const int wv   = tid >> 5;
  const int lane = tid & 31;
  const int l = lane & 15, h = lane >> 4;
  const int wr = wv & 7;        // wave row block
  const int ch = wv >> 3;       // wave column half (0/1)

  // P loader: one fixed row per thread -> hoist reciprocal
  const int prow = tid >> 2;
  const int pq   = tid & 3;
  const float rv = rinv[qblock + prow];

  // V loader: half a column's 16 pair-dwords per thread (contiguous copies)
  const int vcol  = tid >> 1;
  const int vhalf = tid & 1;

  const v8f vzero = {0.f,0.f,0.f,0.f,0.f,0.f,0.f,0.f};
  v8f acc[8];
#pragma unroll
  for (int t = 0; t < 8; ++t) acc[t] = vzero;

  for (int kk = 0; kk < NK; kk += 32) {
    {  // V tile: pure async copy, already B-fragment pair-packed
      const u32* gv = VT + ((size_t)(kk >> 5) * 256 + vcol) * 16 + vhalf * 8;
      u32* dv = vsT + vcol * 20 + vhalf * 8;
      async_b128(dv,     gv);
      async_b128(dv + 4, gv + 4);
    }
    {  // P tile: read raw exp, normalize, write back, pack bf16 into LDS
      float* pp = attn + (size_t)(qblock + prow) * NK + kk + pq * 8;
      float4 p0 = ((const float4*)pp)[0];
      float4 p1 = ((const float4*)pp)[1];
      p0.x *= rv; p0.y *= rv; p0.z *= rv; p0.w *= rv;
      p1.x *= rv; p1.y *= rv; p1.z *= rv; p1.w *= rv;
      ((float4*)pp)[0] = p0;            // normalized weights in place
      ((float4*)pp)[1] = p1;
      u32* pd = ps + prow * 17 + pq * 4;
      pd[0] = pack_bf16(p0.x, p0.y); pd[1] = pack_bf16(p0.z, p0.w);
      pd[2] = pack_bf16(p1.x, p1.y); pd[3] = pack_bf16(p1.z, p1.w);
    }
    async_wait0();
    __syncthreads();

    v8u au;
    const u32* ap = ps + (16 * wr + l) * 17 + h * 8;
    au[0] = ap[0]; au[1] = ap[1]; au[2] = ap[2]; au[3] = ap[3];
    au[4] = ap[4]; au[5] = ap[5]; au[6] = ap[6]; au[7] = ap[7];
    v16bf a = __builtin_bit_cast(v16bf, au);

#pragma unroll
    for (int t = 0; t < 8; ++t) {
      const u32* bp = vsT + (ch * 128 + 16 * t + l) * 20 + h * 8;
      v8u bu;
      bu[0] = bp[0]; bu[1] = bp[1]; bu[2] = bp[2]; bu[3] = bp[3];
      bu[4] = bp[4]; bu[5] = bp[5]; bu[6] = bp[6]; bu[7] = bp[7];
      v16bf b = __builtin_bit_cast(v16bf, bu);
      acc[t] = __builtin_amdgcn_wmma_f32_16x16x32_bf16(
          false, a, false, b, (short)0, acc[t], false, false);
    }
    __syncthreads();
  }

#pragma unroll
  for (int t = 0; t < 8; ++t) {
#pragma unroll
    for (int r = 0; r < 8; ++r) {
      const int qrow = qblock + 16 * wr + 8 * h + r;
      const int col  = ch * 128 + 16 * t + l;
      X[(size_t)qrow * DVV + col] = acc[t][r];
    }
  }
}

// ---------------------------------------------------------------------------
extern "C" void kernel_launch(void* const* d_in, const int* in_sizes, int n_in,
                              void* d_out, int out_size, void* d_ws, size_t ws_size,
                              hipStream_t stream) {
  (void)in_sizes; (void)n_in; (void)out_size; (void)ws_size;
  const float* Q = (const float*)d_in[0];
  const float* K = (const float*)d_in[1];
  const float* V = (const float*)d_in[2];
  float* X    = (float*)d_out;
  float* attn = X + (size_t)NQ * DVV;              // second tuple output

  float* rinv    = (float*)d_ws;                   // NQ floats (1/rowsum)
  float* partial = rinv + NQ;                      // (NK/128)*NQ floats
  u32*   QB      = (u32*)(partial + (size_t)(NK / 128) * NQ);  // NQ*DD/2 dwords
  u32*   KB      = QB + (size_t)NQ * (DD / 2);     // NK*DD/2 dwords
  u32*   VT      = KB + (size_t)NK * (DD / 2);     // NK*DVV/2 dwords

  pack_rows_kernel<<<(NQ * (DD / 2)) / 256, 256, 0, stream>>>(Q, QB);
  pack_rows_kernel<<<(NK * (DD / 2)) / 256, 256, 0, stream>>>(K, KB);
  pack_vt_kernel<<<NK / 32, 256, 0, stream>>>(V, VT);

  scores_kernel<<<dim3(NK / 128, NQ / 128), 256, 0, stream>>>(QB, KB, attn, partial);
  rowsum_kernel<<<NQ / 256, 256, 0, stream>>>(partial, rinv);
  pv_kernel<<<NQ / 128, 512, 0, stream>>>(VT, attn, rinv, X);
}
